// ContactGraspNet_2508260901649
// MI455X (gfx1250) — matrix operand order repaired and checked by reading, not compiled, not run on hardware
//
#include <hip/hip_runtime.h>
#include <hip/hip_bf16.h>
#include <math.h>

typedef _Float16 v16h __attribute__((ext_vector_type(16)));
typedef _Float16 v8h  __attribute__((ext_vector_type(8)));
typedef float    v8f  __attribute__((ext_vector_type(8)));

#define BN_INV_F 0.9999950000374997f
#define FPS_T 256

// ---------------------------------------------------------------------------
// Weight prep: fold BN (g, bt, BN_INV) into weight + bias, transpose to
// f16 [Npad][Kpad] so the GEMM's B-fragment loads are contiguous in K.
// ---------------------------------------------------------------------------
__global__ void cgn_prep_weights(const float* __restrict__ W, const float* __restrict__ b,
                                 const float* __restrict__ bt, const float* __restrict__ g,
                                 int cin, int cout, int Kpad, int Npad,
                                 _Float16* __restrict__ Wt, float* __restrict__ biasOut, int hasBN)
{
    int n = blockIdx.x * blockDim.x + threadIdx.x;
    if (n >= Npad) return;
    float scale = 1.f, bb = 0.f;
    if (n < cout) {
        scale = hasBN ? g[n] * BN_INV_F : 1.f;
        bb = b[n] * scale + (hasBN ? bt[n] : 0.f);
    }
    biasOut[n] = bb;
    _Float16* row = Wt + (long)n * Kpad;
    for (int kk = 0; kk < Kpad; ++kk) {
        float w = (n < cout && kk < cin) ? W[(long)kk * cout + n] * scale : 0.f;
        row[kk] = (_Float16)w;
    }
}

// ---------------------------------------------------------------------------
// WMMA GEMM:  out[M,Npad] = act(A[M,Kpad] @ Wt^T + bias)
// A: f16 row-major [M,Kpad]; Wt: f16 [Npad][Kpad] (K contiguous); f32 accum.
// NW = N-tiles per wave (2 => 16x32 patch, one A fragment feeds two WMMAs).
// Ping-pong unroll-by-2: two fragment register sets alternate by loop
// structure (no rotation moves); loads for the next 32-K phase are issued
// before the WMMAs of the current phase.
//
// A-fragment (16x32 f16): lane holds row M=lane%16; halves i=0..7 -> K=hilo*8+i,
// i=8..15 -> K=16+hilo*8+(i-8)  (hilo = lane>=16)  => two b128 loads.
// B-fragment (32x16 f16): lane holds col N=lane%16; same K mapping as A
//   => two b128 loads from the transposed weight row.
// C/D (f32 16x16): element v -> M = v + hilo*8, N = lane%16.
// ---------------------------------------------------------------------------
template <int NW>
__global__ void cgn_gemm_wmma(const _Float16* __restrict__ A, const _Float16* __restrict__ Wt,
                              const float* __restrict__ bias, void* __restrict__ out,
                              int M, int Kpad, int Npad, int relu, int out_f32)
{
    union Frag { v16h v; v8h h[2]; };

    int lane = threadIdx.x & 31;
    int wave = threadIdx.x >> 5;
    int tilesN = (Npad >> 4) / NW;
    long tile = (long)blockIdx.x * (blockDim.x >> 5) + wave;
    long totalTiles = (long)(M >> 4) * tilesN;
    if (tile >= totalTiles) return;
    int  n0 = (int)(tile % tilesN) * (16 * NW);
    long m0 = (tile / tilesN) << 4;
    int row  = lane & 15;
    int hilo = lane >> 4;

    const _Float16* arow  = A  + (m0 + row) * (long)Kpad;
    const _Float16* brow0 = Wt + (long)(n0 + row) * Kpad;
    const _Float16* brow1 = Wt + (long)(n0 + 16 + row) * Kpad;

    auto loadA = [&](Frag& f, int k) {
        f.h[0] = *(const v8h*)(arow + k + hilo * 8);
        f.h[1] = *(const v8h*)(arow + k + 16 + hilo * 8);
    };
    auto loadB = [&](Frag& f, const _Float16* br, int k) {
        f.h[0] = *(const v8h*)(br + k + hilo * 8);
        f.h[1] = *(const v8h*)(br + k + 16 + hilo * 8);
    };

    Frag a0, a1, p00, p01, p10, p11;   // pXY: B fragment, X = n-tile, Y = phase
    v8f c0 = {}, c1 = {};

    loadA(a0, 0);
    loadB(p00, brow0, 0);
    if (NW == 2) loadB(p10, brow1, 0);

    int k0 = 0;
    while (true) {
        int k1 = k0 + 32;
        if (k1 < Kpad) {                       // issue phase-1 loads early
            loadA(a1, k1);
            loadB(p01, brow0, k1);
            if (NW == 2) loadB(p11, brow1, k1);
            __builtin_prefetch(arow + k1 + 32, 0, 1);
        }
        c0 = __builtin_amdgcn_wmma_f32_16x16x32_f16(false, a0.v, false, p00.v, (short)0, c0,
                                                    false, false);
        if (NW == 2)
            c1 = __builtin_amdgcn_wmma_f32_16x16x32_f16(false, a0.v, false, p10.v, (short)0, c1,
                                                        false, false);
        if (k1 >= Kpad) break;

        int k2 = k0 + 64;
        if (k2 < Kpad) {                       // issue phase-0 loads early
            loadA(a0, k2);
            loadB(p00, brow0, k2);
            if (NW == 2) loadB(p10, brow1, k2);
            __builtin_prefetch(arow + k2 + 32, 0, 1);
        }
        c0 = __builtin_amdgcn_wmma_f32_16x16x32_f16(false, a1.v, false, p01.v, (short)0, c0,
                                                    false, false);
        if (NW == 2)
            c1 = __builtin_amdgcn_wmma_f32_16x16x32_f16(false, a1.v, false, p11.v, (short)0, c1,
                                                        false, false);
        if (k2 >= Kpad) break;
        k0 = k2;
    }

    float bv0 = bias[n0 + row];
    float bv1 = (NW == 2) ? bias[n0 + 16 + row] : 0.f;
#pragma unroll
    for (int v = 0; v < 8; ++v) {
        long orow = m0 + v + hilo * 8;
        float x0 = c0[v] + bv0;
        if (relu) x0 = fmaxf(x0, 0.f);
        if (out_f32) ((float*)out)[orow * Npad + n0 + row] = x0;
        else         ((_Float16*)out)[orow * Npad + n0 + row] = (_Float16)x0;
        if (NW == 2) {
            float x1 = c1[v] + bv1;
            if (relu) x1 = fmaxf(x1, 0.f);
            if (out_f32) ((float*)out)[orow * Npad + n0 + 16 + row] = x1;
            else         ((_Float16*)out)[orow * Npad + n0 + 16 + row] = (_Float16)x1;
        }
    }
}

// ---------------------------------------------------------------------------
// Farthest point sampling: one block per batch; dist[] in global scratch,
// block-wide argmax reduction per step (first-occurrence tie-break like argmax).
// ---------------------------------------------------------------------------
__global__ void cgn_fps(const float* __restrict__ xyz, int N, int np,
                        int* __restrict__ fidx, float* __restrict__ newxyz,
                        float* __restrict__ dist)
{
    int b = blockIdx.x;
    const float* x = xyz + (long)b * N * 3;
    float* d = dist + (long)b * N;
    __shared__ float smax[FPS_T];
    __shared__ int   simx[FPS_T];
    int t = threadIdx.x;
    for (int i = t; i < N; i += FPS_T) d[i] = 1e10f;
    __syncthreads();
    int far = 0;
    for (int j = 0; j < np; ++j) {
        if (t == 0) {
            fidx[(long)b * np + j] = far;
            newxyz[((long)b * np + j) * 3 + 0] = x[far * 3 + 0];
            newxyz[((long)b * np + j) * 3 + 1] = x[far * 3 + 1];
            newxyz[((long)b * np + j) * 3 + 2] = x[far * 3 + 2];
        }
        float cx = x[far * 3 + 0], cy = x[far * 3 + 1], cz = x[far * 3 + 2];
        float best = -1.f; int bi = 0x7fffffff;
        for (int i = t; i < N; i += FPS_T) {
            float dx = x[i * 3 + 0] - cx, dy = x[i * 3 + 1] - cy, dz = x[i * 3 + 2] - cz;
            float nd = fminf(d[i], dx * dx + dy * dy + dz * dz);
            d[i] = nd;
            if (nd > best) { best = nd; bi = i; }
        }
        smax[t] = best; simx[t] = bi;
        __syncthreads();
        for (int s = FPS_T >> 1; s > 0; s >>= 1) {
            if (t < s) {
                if (smax[t + s] > smax[t] ||
                    (smax[t + s] == smax[t] && simx[t + s] < simx[t])) {
                    smax[t] = smax[t + s]; simx[t] = simx[t + s];
                }
            }
            __syncthreads();
        }
        far = simx[0];
        __syncthreads();
    }
}

// ---------------------------------------------------------------------------
// Ball query: one wave per query center; wave32 ballot stream-compaction picks
// the k smallest in-range indices in order, pads with the first hit.
// ---------------------------------------------------------------------------
__global__ void cgn_ball_query(const float* __restrict__ xyz, const float* __restrict__ newxyz,
                               int N, int np, float r2, int k, int* __restrict__ gi, int B_)
{
    int lane = threadIdx.x & 31;
    long wid = (long)blockIdx.x * (blockDim.x >> 5) + (threadIdx.x >> 5);
    if (wid >= (long)B_ * np) return;
    int b = (int)(wid / np);
    const float* x = xyz + (long)b * N * 3;
    const float* c = newxyz + wid * 3;
    float cx = c[0], cy = c[1], cz = c[2];
    int* g = gi + wid * k;
    int cnt = 0, first = 0, haveFirst = 0;
    for (int base = 0; base < N && cnt < k; base += 32) {
        int i = base + lane;
        bool in = false;
        if (i < N) {
            float dx = x[i * 3 + 0] - cx, dy = x[i * 3 + 1] - cy, dz = x[i * 3 + 2] - cz;
            in = (dx * dx + dy * dy + dz * dz) <= r2;
        }
        unsigned int mask = __builtin_amdgcn_ballot_w32(in);
        if (!haveFirst && mask) { first = base + __builtin_ctz(mask); haveFirst = 1; }
        if (in) {
            int slot = cnt + __popc(mask & ((1u << lane) - 1u));
            if (slot < k) g[slot] = i;
        }
        cnt += __popc(mask);
    }
    if (cnt > k) cnt = k;
    for (int s = cnt + lane; s < k; s += 32) g[s] = first;
}

// Build grouped-feature A matrix: [points(C) , xyz-center(3) , zero pad] as f16.
__global__ void cgn_group_gather(const float* __restrict__ xyz, const _Float16* __restrict__ pts,
                                 const float* __restrict__ newxyz, const int* __restrict__ gi,
                                 int B_, int N, int np, int k, int C, int Kpad,
                                 _Float16* __restrict__ A)
{
    long r = (long)blockIdx.x * blockDim.x + threadIdx.x;
    long total = (long)B_ * np * k;
    if (r >= total) return;
    long cj = r / k;
    int j = (int)(cj % np);
    int b = (int)(cj / np);
    int pi = gi[r];
    _Float16* a = A + r * (long)Kpad;
    int c = 0;
    if (pts) {
        const _Float16* p = pts + ((long)b * N + pi) * C;
        for (; c < C; ++c) a[c] = p[c];
    }
    const float* xp = xyz + ((long)b * N + pi) * 3;
    const float* cp = newxyz + ((long)b * np + j) * 3;
    a[c++] = (_Float16)(xp[0] - cp[0]);
    a[c++] = (_Float16)(xp[1] - cp[1]);
    a[c++] = (_Float16)(xp[2] - cp[2]);
    for (; c < Kpad; ++c) a[c] = (_Float16)0.f;
}

// Max over nsample group members; writes into a concat slot of the SA output.
__global__ void cgn_maxpool(const _Float16* __restrict__ in, _Float16* __restrict__ out,
                            long rows, int k, int Nstride, int Ctot, int coff, int Cuse)
{
    long idx = (long)blockIdx.x * blockDim.x + threadIdx.x;
    long total = rows * Cuse;
    if (idx >= total) return;
    int c = (int)(idx % Cuse);
    long rj = idx / Cuse;
    const _Float16* p = in + (rj * k) * (long)Nstride + c;
    float m = -3.0e38f;
    for (int s = 0; s < k; ++s) m = fmaxf(m, (float)p[(long)s * Nstride]);
    out[rj * Ctot + coff + c] = (_Float16)m;
}

// Global max over n points per batch (SA4).
__global__ void cgn_gmax(const _Float16* __restrict__ in, _Float16* __restrict__ out,
                         int B_, int n, int C)
{
    int idx = blockIdx.x * blockDim.x + threadIdx.x;
    if (idx >= B_ * C) return;
    int c = idx % C, b = idx / C;
    const _Float16* p = in + ((long)b * n) * C + c;
    float m = -3.0e38f;
    for (int s = 0; s < n; ++s) m = fmaxf(m, (float)p[(long)s * C]);
    out[(long)b * C + c] = (_Float16)m;
}

// SA4 input: [xyz(3) , feat(C) , pad].
__global__ void cgn_concat_xyz_feat(const float* __restrict__ xyz, const _Float16* __restrict__ feat,
                                    int B_, int n, int C, int Kpad, _Float16* __restrict__ A)
{
    long r = (long)blockIdx.x * blockDim.x + threadIdx.x;
    if (r >= (long)B_ * n) return;
    _Float16* a = A + r * (long)Kpad;
    const float* x = xyz + r * 3;
    a[0] = (_Float16)x[0]; a[1] = (_Float16)x[1]; a[2] = (_Float16)x[2];
    const _Float16* f = feat + r * (long)C;
    for (int c = 0; c < C; ++c) a[3 + c] = f[c];
    for (int c = 3 + C; c < Kpad; ++c) a[c] = (_Float16)0.f;
}

// FP3 input: [per-point feat(C1) , broadcast global feat(C2)].
__global__ void cgn_concat_feat_global(const _Float16* __restrict__ p1, const _Float16* __restrict__ g,
                                       int B_, int n, int C1, int C2, _Float16* __restrict__ A)
{
    long r = (long)blockIdx.x * blockDim.x + threadIdx.x;
    if (r >= (long)B_ * n) return;
    int b = (int)(r / n);
    _Float16* a = A + r * (long)(C1 + C2);
    const _Float16* q = p1 + r * (long)C1;
    for (int c = 0; c < C1; ++c) a[c] = q[c];
    const _Float16* gg = g + (long)b * C2;
    for (int c = 0; c < C2; ++c) a[C1 + c] = gg[c];
}

// FP interpolation: brute-force 3-NN + inverse-distance weights, concat
// [points1(C1) , interp(C2)] into the next MLP's A matrix.
__global__ void cgn_interp_concat(const float* __restrict__ xyz1, const float* __restrict__ xyz2,
                                  const _Float16* __restrict__ p1, const _Float16* __restrict__ p2,
                                  int B_, int n1, int n2, int C1, int C2, _Float16* __restrict__ A)
{
    long r = (long)blockIdx.x * blockDim.x + threadIdx.x;
    if (r >= (long)B_ * n1) return;
    int b = (int)(r / n1);
    const float* q = xyz1 + r * 3;
    const float* X = xyz2 + (long)b * n2 * 3;
    float d0 = 3e38f, d1 = 3e38f, d2 = 3e38f;
    int i0 = 0, i1 = 0, i2 = 0;
    for (int t = 0; t < n2; ++t) {
        float dx = X[t * 3 + 0] - q[0], dy = X[t * 3 + 1] - q[1], dz = X[t * 3 + 2] - q[2];
        float dd = dx * dx + dy * dy + dz * dz;
        if (dd < d0)      { d2 = d1; i2 = i1; d1 = d0; i1 = i0; d0 = dd; i0 = t; }
        else if (dd < d1) { d2 = d1; i2 = i1; d1 = dd; i1 = t; }
        else if (dd < d2) { d2 = dd; i2 = t; }
    }
    float w0 = 1.f / (d0 + 1e-8f), w1 = 1.f / (d1 + 1e-8f), w2 = 1.f / (d2 + 1e-8f);
    float ws = w0 + w1 + w2; w0 /= ws; w1 /= ws; w2 /= ws;
    _Float16* a = A + r * (long)(C1 + C2);
    const _Float16* q1 = p1 + r * (long)C1;
    for (int c = 0; c < C1; ++c) a[c] = q1[c];
    const _Float16* f0 = p2 + ((long)b * n2 + i0) * C2;
    const _Float16* f1 = p2 + ((long)b * n2 + i1) * C2;
    const _Float16* f2 = p2 + ((long)b * n2 + i2) * C2;
    for (int c = 0; c < C2; ++c)
        a[C1 + c] = (_Float16)(w0 * (float)f0[c] + w1 * (float)f1[c] + w2 * (float)f2[c]);
}

// Final assembly: normalize dir, orthonormalize app, cross product, 4x4 grasp,
// sigmoid score, width bin lookup, wlog transpose. Outputs concatenated flat.
__global__ void cgn_finalize(const float* __restrict__ dirh, const float* __restrict__ apph,
                             const float* __restrict__ widh, const float* __restrict__ segh,
                             const float* __restrict__ l1xyz, float* __restrict__ out,
                             int B_, int n)
{
    long r = (long)blockIdx.x * blockDim.x + threadIdx.x;
    long total = (long)B_ * n;
    if (r >= total) return;
    int b = (int)(r / n), i = (int)(r % n);
    long OFF_S = total * 16;
    long OFF_P = OFF_S + total;
    long OFF_W = OFF_P + total * 3;
    long OFF_L = OFF_W + total;
    const float* dv = dirh + r * 16;
    float dx = dv[0], dy = dv[1], dz = dv[2];
    float nn = fmaxf(sqrtf(dx * dx + dy * dy + dz * dz), 1e-12f);
    dx /= nn; dy /= nn; dz /= nn;
    const float* av = apph + r * 16;
    float ax = av[0], ay = av[1], az = av[2];
    float dot = ax * dx + ay * dy + az * dz;
    ax -= dot * dx; ay -= dot * dy; az -= dot * dz;
    float an = fmaxf(sqrtf(ax * ax + ay * ay + az * az), 1e-12f);
    ax /= an; ay /= an; az /= an;
    float cxv = ay * dz - az * dy;
    float cyv = az * dx - ax * dz;
    float czv = ax * dy - ay * dx;
    const float* xz = l1xyz + r * 3;
    float* gsp = out + r * 16;
    gsp[0] = dx;  gsp[1] = cxv;  gsp[2] = ax;  gsp[3] = xz[0];
    gsp[4] = dy;  gsp[5] = cyv;  gsp[6] = ay;  gsp[7] = xz[1];
    gsp[8] = dz;  gsp[9] = czv;  gsp[10] = az; gsp[11] = xz[2];
    gsp[12] = 0.f; gsp[13] = 0.f; gsp[14] = 0.f; gsp[15] = 1.f;
    float sg = segh[r * 16];
    out[OFF_S + r] = 1.f / (1.f + __expf(-sg));
    out[OFF_P + r * 3 + 0] = xz[0];
    out[OFF_P + r * 3 + 1] = xz[1];
    out[OFF_P + r * 3 + 2] = xz[2];
    const float* wl = widh + r * 16;
    int am = 0; float bm = wl[0];
    for (int t = 1; t < 10; ++t) if (wl[t] > bm) { bm = wl[t]; am = t; }
    const float BINS[10] = {
        0.00397217665f, 0.01191652770f, 0.01986087650f, 0.02780522525f,
        0.03574957400f, 0.04369392280f, 0.05163827160f, 0.05958262040f,
        0.06752696915f, 0.075f };
    out[OFF_W + r] = BINS[am];
    for (int t = 0; t < 10; ++t)
        out[OFF_L + ((long)b * 10 + t) * n + i] = wl[t];
}

// ---------------------------------------------------------------------------
// Host orchestration
// ---------------------------------------------------------------------------
extern "C" void kernel_launch(void* const* d_in, const int* in_sizes, int n_in,
                              void* d_out, int out_size, void* d_ws, size_t ws_size,
                              hipStream_t stream)
{
    (void)in_sizes; (void)n_in; (void)out_size; (void)ws_size;
    const int B_ = 2, NPTS = 20000;

    // ---- input leaves (JAX pytree order: dict keys sorted at every level) ----
    int cur = 0;
    const float* pc = (const float*)d_in[cur++];

    struct Lin { const float *W, *b, *bt, *g; };
    struct HeadP { Lin l1, l2; };
    auto takeBN = [&](void) -> Lin {
        Lin L; L.W = (const float*)d_in[cur++]; L.b = (const float*)d_in[cur++];
        L.bt = (const float*)d_in[cur++]; L.g = (const float*)d_in[cur++]; return L;
    };
    auto takeNoBN = [&](void) -> Lin {
        Lin L; L.W = (const float*)d_in[cur++]; L.b = (const float*)d_in[cur++];
        L.bt = nullptr; L.g = nullptr; return L;
    };
    auto takeHead = [&](void) -> HeadP { HeadP h; h.l1 = takeBN(); h.l2 = takeNoBN(); return h; };

    HeadP appH = takeHead();
    HeadP dirH = takeHead();
    Lin fp1L[2] = { takeBN(), takeBN() };
    Lin fp2L[2] = { takeBN(), takeBN() };
    Lin fp3L[2] = { takeBN(), takeBN() };
    Lin sa1L[3][3], sa2L[3][3], sa3L[3][3];
    for (int s = 0; s < 3; ++s) for (int l = 0; l < 3; ++l) sa1L[s][l] = takeBN();
    for (int s = 0; s < 3; ++s) for (int l = 0; l < 3; ++l) sa2L[s][l] = takeBN();
    for (int s = 0; s < 3; ++s) for (int l = 0; l < 3; ++l) sa3L[s][l] = takeBN();
    Lin sa4L[3] = { takeBN(), takeBN(), takeBN() };
    HeadP segH = takeHead();
    HeadP widH = takeHead();

    // ---- workspace bump allocator ----
    char* wsB = (char*)d_ws;
    size_t wo = 0;
    auto bump = [&](size_t bytes) -> void* {
        size_t a = (wo + 255) & ~(size_t)255;
        void* p = wsB + a; wo = a + bytes; return p;
    };
    auto pad16 = [](int v) { return (v + 15) & ~15; };
    auto pad32 = [](int v) { return (v + 31) & ~31; };

    // ---- prep all linear layers (BN folded, transposed f16) ----
    struct PLayer { _Float16* Wt; float* bias; int cin, cout, Kpad, Npad; };
    auto prep = [&](const Lin& L, int cin, int cout) -> PLayer {
        PLayer P; P.cin = cin; P.cout = cout; P.Kpad = pad32(cin); P.Npad = pad16(cout);
        P.Wt = (_Float16*)bump((size_t)P.Npad * P.Kpad * sizeof(_Float16));
        P.bias = (float*)bump((size_t)P.Npad * sizeof(float));
        cgn_prep_weights<<<dim3((unsigned)((P.Npad + 63) / 64)), dim3(64), 0, stream>>>(
            L.W, L.b, L.bt, L.g, cin, cout, P.Kpad, P.Npad, P.Wt, P.bias, L.g ? 1 : 0);
        return P;
    };

    const int sa1c[3][4] = {{3,32,32,64},{3,64,64,128},{3,64,96,128}};
    const int sa2c[3][4] = {{323,64,64,128},{323,128,128,256},{323,128,128,256}};
    const int sa3c[3][4] = {{643,64,64,128},{643,128,128,256},{643,128,128,256}};
    const int sa4c[4]    = {643,128,256,1024};

    PLayer sa1P[3][3], sa2P[3][3], sa3P[3][3], sa4P[3];
    for (int s = 0; s < 3; ++s) for (int l = 0; l < 3; ++l)
        sa1P[s][l] = prep(sa1L[s][l], sa1c[s][l], sa1c[s][l + 1]);
    for (int s = 0; s < 3; ++s) for (int l = 0; l < 3; ++l)
        sa2P[s][l] = prep(sa2L[s][l], sa2c[s][l], sa2c[s][l + 1]);
    for (int s = 0; s < 3; ++s) for (int l = 0; l < 3; ++l)
        sa3P[s][l] = prep(sa3L[s][l], sa3c[s][l], sa3c[s][l + 1]);
    for (int l = 0; l < 3; ++l) sa4P[l] = prep(sa4L[l], sa4c[l], sa4c[l + 1]);
    PLayer fp3P[2] = { prep(fp3L[0], 1664, 128), prep(fp3L[1], 128, 128) };
    PLayer fp2P[2] = { prep(fp2L[0],  768, 128), prep(fp2L[1], 128, 128) };
    PLayer fp1P[2] = { prep(fp1L[0],  448, 128), prep(fp1L[1], 128, 128) };
    struct PHead { PLayer l1, l2; };
    PHead dirP = { prep(dirH.l1, 128, 128), prep(dirH.l2, 128, 3) };
    PHead appP = { prep(appH.l1, 128, 128), prep(appH.l2, 128, 3) };
    PHead widP = { prep(widH.l1, 128, 128), prep(widH.l2, 128, 10) };
    PHead segP = { prep(segH.l1, 128, 128), prep(segH.l2, 128, 1) };

    // ---- activation / scratch buffers ----
    float* dist   = (float*)bump((size_t)B_ * NPTS * sizeof(float));
    int*   fidx   = (int*)bump((size_t)B_ * 2048 * sizeof(int));
    int*   gi     = (int*)bump((size_t)B_ * 2048 * 128 * sizeof(int));
    float* l1xyz  = (float*)bump((size_t)B_ * 2048 * 3 * sizeof(float));
    float* l2xyz  = (float*)bump((size_t)B_ * 512 * 3 * sizeof(float));
    float* l3xyz  = (float*)bump((size_t)B_ * 128 * 3 * sizeof(float));
    _Float16* l1p  = (_Float16*)bump((size_t)B_ * 2048 * 320 * 2);
    _Float16* l2p  = (_Float16*)bump((size_t)B_ * 512 * 640 * 2);
    _Float16* l3p  = (_Float16*)bump((size_t)B_ * 128 * 640 * 2);
    _Float16* l4p  = (_Float16*)bump((size_t)B_ * 1024 * 2);
    _Float16* fp3o = (_Float16*)bump((size_t)B_ * 128 * 128 * 2);
    _Float16* fp2o = (_Float16*)bump((size_t)B_ * 512 * 128 * 2);
    _Float16* fp1o = (_Float16*)bump((size_t)B_ * 2048 * 128 * 2);
    float* dirO = (float*)bump((size_t)B_ * 2048 * 16 * sizeof(float));
    float* appO = (float*)bump((size_t)B_ * 2048 * 16 * sizeof(float));
    float* widO = (float*)bump((size_t)B_ * 2048 * 16 * sizeof(float));
    float* segO = (float*)bump((size_t)B_ * 2048 * 16 * sizeof(float));
    _Float16* bufA = (_Float16*)bump((size_t)131072 * 352 * 2); // largest A matrix (SA2 s3)
    _Float16* bufB = (_Float16*)bump((size_t)524288 *  64 * 2); // largest even-layer/final out
    _Float16* bufC = (_Float16*)bump((size_t)524288 *  32 * 2); // largest odd-layer out

    auto gemm = [&](const _Float16* A, const PLayer& P, void* out, int M, int relu, int f32) {
        if ((P.Npad & 31) == 0) {
            long tiles = (long)(M / 16) * (P.Npad / 32);
            unsigned blocks = (unsigned)((tiles + 7) / 8);
            cgn_gemm_wmma<2><<<dim3(blocks), dim3(256), 0, stream>>>(
                A, P.Wt, P.bias, out, M, P.Kpad, P.Npad, relu, f32);
        } else {
            long tiles = (long)(M / 16) * (P.Npad / 16);
            unsigned blocks = (unsigned)((tiles + 7) / 8);
            cgn_gemm_wmma<1><<<dim3(blocks), dim3(256), 0, stream>>>(
                A, P.Wt, P.bias, out, M, P.Kpad, P.Npad, relu, f32);
        }
    };

    // ---- set abstraction (multi-scale grouping) ----
    auto runSA = [&](const float* xyzIn, int Nin, const _Float16* ptsIn, int C,
                     int np, const float* radii, const int* ks, PLayer (*Ls)[3],
                     float* xyzOut, _Float16* pOut, int Ctot) {
        cgn_fps<<<dim3((unsigned)B_), dim3(FPS_T), 0, stream>>>(xyzIn, Nin, np, fidx, xyzOut, dist);
        int coff = 0;
        for (int s = 0; s < 3; ++s) {
            int k = ks[s];
            float r2 = radii[s] * radii[s];
            long waves = (long)B_ * np;
            cgn_ball_query<<<dim3((unsigned)((waves + 7) / 8)), dim3(256), 0, stream>>>(
                xyzIn, xyzOut, Nin, np, r2, k, gi, B_);
            long rows = (long)B_ * np * k;
            int Kp = pad32(C + 3);
            cgn_group_gather<<<dim3((unsigned)((rows + 255) / 256)), dim3(256), 0, stream>>>(
                xyzIn, ptsIn, xyzOut, gi, B_, Nin, np, k, C, Kp, bufA);
            gemm(bufA, Ls[s][0], bufB, (int)rows, 1, 0);
            gemm(bufB, Ls[s][1], bufC, (int)rows, 1, 0);
            gemm(bufC, Ls[s][2], bufB, (int)rows, 1, 0);
            int cl = Ls[s][2].cout;
            long total = (long)B_ * np * cl;
            cgn_maxpool<<<dim3((unsigned)((total + 255) / 256)), dim3(256), 0, stream>>>(
                bufB, pOut, (long)B_ * np, k, Ls[s][2].Npad, Ctot, coff, cl);
            coff += cl;
        }
    };

    const float sa1r[3] = {0.02f, 0.04f, 0.08f}; const int sa1k[3] = {32, 64, 128};
    const float sa2r[3] = {0.04f, 0.08f, 0.16f}; const int sa2k[3] = {64, 64, 128};
    const float sa3r[3] = {0.08f, 0.16f, 0.32f}; const int sa3k[3] = {64, 64, 128};

    runSA(pc,    NPTS, nullptr, 0,   2048, sa1r, sa1k, sa1P, l1xyz, l1p, 320);
    runSA(l1xyz, 2048, l1p,     320,  512, sa2r, sa2k, sa2P, l2xyz, l2p, 640);
    runSA(l2xyz,  512, l2p,     640,  128, sa3r, sa3k, sa3P, l3xyz, l3p, 640);

    // ---- SA4 (global) ----
    {
        long rows = (long)B_ * 128;
        cgn_concat_xyz_feat<<<dim3((unsigned)((rows + 255) / 256)), dim3(256), 0, stream>>>(
            l3xyz, l3p, B_, 128, 640, 672, bufA);
        gemm(bufA, sa4P[0], bufB, (int)rows, 1, 0);
        gemm(bufB, sa4P[1], bufC, (int)rows, 1, 0);
        gemm(bufC, sa4P[2], bufB, (int)rows, 1, 0);
        cgn_gmax<<<dim3((unsigned)((B_ * 1024 + 255) / 256)), dim3(256), 0, stream>>>(
            bufB, l4p, B_, 128, 1024);
    }
    // ---- FP3 ----
    {
        long rows = (long)B_ * 128;
        cgn_concat_feat_global<<<dim3((unsigned)((rows + 255) / 256)), dim3(256), 0, stream>>>(
            l3p, l4p, B_, 128, 640, 1024, bufA);
        gemm(bufA, fp3P[0], bufB, (int)rows, 1, 0);
        gemm(bufB, fp3P[1], fp3o, (int)rows, 1, 0);
    }
    // ---- FP2 ----
    {
        long rows = (long)B_ * 512;
        cgn_interp_concat<<<dim3((unsigned)((rows + 255) / 256)), dim3(256), 0, stream>>>(
            l2xyz, l3xyz, l2p, fp3o, B_, 512, 128, 640, 128, bufA);
        gemm(bufA, fp2P[0], bufB, (int)rows, 1, 0);
        gemm(bufB, fp2P[1], fp2o, (int)rows, 1, 0);
    }
    // ---- FP1 ----
    {
        long rows = (long)B_ * 2048;
        cgn_interp_concat<<<dim3((unsigned)((rows + 255) / 256)), dim3(256), 0, stream>>>(
            l1xyz, l2xyz, l1p, fp2o, B_, 2048, 512, 320, 128, bufA);
        gemm(bufA, fp1P[0], bufB, (int)rows, 1, 0);
        gemm(bufB, fp1P[1], fp1o, (int)rows, 1, 0);
    }
    // ---- heads ----
    {
        int M = B_ * 2048;
        gemm(fp1o, dirP.l1, bufC, M, 1, 0); gemm(bufC, dirP.l2, dirO, M, 0, 1);
        gemm(fp1o, appP.l1, bufC, M, 1, 0); gemm(bufC, appP.l2, appO, M, 0, 1);
        gemm(fp1o, widP.l1, bufC, M, 1, 0); gemm(bufC, widP.l2, widO, M, 0, 1);
        gemm(fp1o, segP.l1, bufC, M, 1, 0); gemm(bufC, segP.l2, segO, M, 0, 1);
    }
    cgn_finalize<<<dim3((unsigned)((B_ * 2048 + 255) / 256)), dim3(256), 0, stream>>>(
        dirO, appO, widO, segO, l1xyz, (float*)d_out, B_, 2048);
}
